// SeqModelFused_16569983828595
// MI455X (gfx1250) — matrix operand order repaired and checked by reading, not compiled, not run on hardware
//
#include <hip/hip_runtime.h>

typedef __attribute__((ext_vector_type(16))) __bf16 v16bf;
typedef __attribute__((ext_vector_type(8)))  __bf16 v8bf;
typedef __attribute__((ext_vector_type(4)))  __bf16 v4bf;
typedef __attribute__((ext_vector_type(8)))  float  v8f;

#define TILE_M 256          // block tile rows
#define TILE_N 128          // block tile cols
#define TILE_K 32
#define LDS_STRIDE 40       // 32 + 8 bf16 pad; 80B row stride stays 16B aligned

// ---------------------------------------------------------------------------
// CDNA5 async copy: 16 bytes per lane, global -> LDS, tracked by ASYNCcnt.
// lds operand = wave-relative LDS byte offset (low 32 bits of generic addr).
// ---------------------------------------------------------------------------
__device__ __forceinline__ void async_copy_b128(const void* gptr, void* lptr) {
    const unsigned long long ga = (unsigned long long)(uintptr_t)gptr;
    const unsigned           la = (unsigned)(uintptr_t)lptr;
    asm volatile("global_load_async_to_lds_b128 %0, %1, off"
                 :: "v"(la), "v"(ga) : "memory");
}
__device__ __forceinline__ void wait_asynccnt_le6() {
    asm volatile("s_wait_asynccnt 0x6" ::: "memory");
}
__device__ __forceinline__ void wait_asynccnt_0() {
    asm volatile("s_wait_asynccnt 0x0" ::: "memory");
}

// Stage a ROWSx32 bf16 tile as 16B chunks, ROWS*4/256 chunks per thread.
template <int ROWS>
__device__ __forceinline__ void async_tile(const __bf16* __restrict__ gbase,
                                           size_t row_stride,  // in elements
                                           __bf16 (*lds)[LDS_STRIDE], int tid)
{
    #pragma unroll
    for (int part = 0; part < ROWS / 64; ++part) {
        const int c = tid + part * 256;
        const int r = c >> 2;            // 0..ROWS-1
        const int q = (c & 3) * 8;       // bf16 offset: 0,8,16,24 (16B chunks)
        async_copy_b128(gbase + (size_t)r * row_stride + q, &lds[r][q]);
    }
}

// One K-step of a 64x64 wave tile: 16 fragment ds_loads feeding 16 WMMAs.
__device__ __forceinline__ void compute_step(const __bf16 (*la)[LDS_STRIDE],
                                             const __bf16 (*lb)[LDS_STRIDE],
                                             int wm, int wn, int ln, int lh,
                                             v8f acc[4][4])
{
    // A fragments: 16x32 bf16; lane ln = row, lh picks K-halves
    v16bf af[4];
    #pragma unroll
    for (int i = 0; i < 4; ++i) {
        const int r = wm * 64 + i * 16 + ln;
        v8bf a0 = *(const v8bf*)&la[r][8 * lh];
        v8bf a1 = *(const v8bf*)&la[r][16 + 8 * lh];
        af[i] = __builtin_shufflevector(a0, a1, 0, 1, 2, 3, 4, 5, 6, 7,
                                                8, 9, 10, 11, 12, 13, 14, 15);
    }
    // B fragments: 32x16 bf16; lane ln = column, lh picks K-halves
    v16bf bfr[4];
    #pragma unroll
    for (int j = 0; j < 4; ++j) {
        const int c = wn * 64 + j * 16 + ln;
        v8bf b0 = *(const v8bf*)&lb[c][8 * lh];
        v8bf b1 = *(const v8bf*)&lb[c][16 + 8 * lh];
        bfr[j] = __builtin_shufflevector(b0, b1, 0, 1, 2, 3, 4, 5, 6, 7,
                                                 8, 9, 10, 11, 12, 13, 14, 15);
    }
    #pragma unroll
    for (int i = 0; i < 4; ++i)
        #pragma unroll
        for (int j = 0; j < 4; ++j)
            acc[i][j] = __builtin_amdgcn_wmma_f32_16x16x32_bf16(
                false, af[i], false, bfr[j], (short)0, acc[i][j], false, false);
}

// ---------------------------------------------------------------------------
// C[M,N] = A[M,K] * Bw[N,K]^T   (NT GEMM, bf16 operands, K-contiguous)
// 256x128 block tile, 8 waves x (64x64), double-buffered async LDS staging.
// ---------------------------------------------------------------------------
__global__ __launch_bounds__(256, 1)
void gemm_nt_bf16_wmma(const __bf16* __restrict__ A,
                       const __bf16* __restrict__ Bw,
                       float* __restrict__ C,
                       int M, int N, int K)
{
    __shared__ __align__(16) __bf16 lds_a[2][TILE_M][LDS_STRIDE];  // 2 x 20 KB
    __shared__ __align__(16) __bf16 lds_b[2][TILE_N][LDS_STRIDE];  // 2 x 10 KB

    const int tid  = threadIdx.x;
    const int lane = tid & 31;
    const int wave = tid >> 5;
    const int ln   = lane & 15;
    const int lh   = lane >> 4;
    const int wm   = wave & 3;    // 4 waves over 256 rows (64 each)
    const int wn   = wave >> 2;   // 2 waves over 128 cols (64 each)

    const int m0 = blockIdx.y * TILE_M;
    const int n0 = blockIdx.x * TILE_N;

    const __bf16* Abase = A  + (size_t)m0 * K;
    const __bf16* Bbase = Bw + (size_t)n0 * K;

    v8f acc[4][4] = {};

    // prologue: kick off first tile pair into buffer 0 (6 async ops/thread)
    async_tile<TILE_M>(Abase, (size_t)K, lds_a[0], tid);
    async_tile<TILE_N>(Bbase, (size_t)K, lds_b[0], tid);

    int buf = 0;
    int k0  = 0;
    // steady state: unconditional issue-next -> wait current -> compute
    for (; k0 + TILE_K < K; k0 += TILE_K, buf ^= 1) {
        async_tile<TILE_M>(Abase + (k0 + TILE_K), (size_t)K, lds_a[buf ^ 1], tid);
        async_tile<TILE_N>(Bbase + (k0 + TILE_K), (size_t)K, lds_b[buf ^ 1], tid);
        wait_asynccnt_le6();       // in-order: current tile's 6 ops are done
        __syncthreads();           // all waves' staging of current buf visible
        compute_step(lds_a[buf], lds_b[buf], wm, wn, ln, lh, acc);
        __syncthreads();           // readers done before buf is overwritten
    }
    // epilogue K-step
    wait_asynccnt_0();
    __syncthreads();
    compute_step(lds_a[buf], lds_b[buf], wm, wn, ln, lh, acc);

    // store: v8f elem e -> row = base + 8*lh + e, col = base + ln
    #pragma unroll
    for (int i = 0; i < 4; ++i) {
        #pragma unroll
        for (int j = 0; j < 4; ++j) {
            float* out = C + (size_t)(m0 + wm * 64 + i * 16 + 8 * lh) * N
                           + (n0 + wn * 64 + j * 16 + ln);
            #pragma unroll
            for (int e = 0; e < 8; ++e)
                out[(size_t)e * N] = acc[i][j][e];
        }
    }
}

// ---------------------------------------------------------------------------
// f32 -> bf16 pre-convert (x, W1, W2), 4 elems/thread, fully coalesced
// ---------------------------------------------------------------------------
__global__ __launch_bounds__(256)
void cvt_f32_bf16_kernel(const float* __restrict__ s, __bf16* __restrict__ d, int n)
{
    const int i = (blockIdx.x * blockDim.x + threadIdx.x) * 4;
    if (i >= n) return;
    const float4 v = *(const float4*)(s + i);
    v4bf o;
    o[0] = (__bf16)v.x; o[1] = (__bf16)v.y; o[2] = (__bf16)v.z; o[3] = (__bf16)v.w;
    *(v4bf*)(d + i) = o;
}

// ---------------------------------------------------------------------------
// LIF scan: v' = v + (h - v)/TAU; spike = (v' >= 1); reset-to-zero.
// ---------------------------------------------------------------------------
__global__ __launch_bounds__(256)
void lif_scan_kernel(const float* __restrict__ h, __bf16* __restrict__ spikes,
                     int T, int BN)
{
    const int idx = blockIdx.x * blockDim.x + threadIdx.x;
    if (idx >= BN) return;
    float v = 0.0f;
    for (int t = 0; t < T; ++t) {
        const float ht = h[(size_t)t * BN + idx];
        const float hv = v + (ht - v) * 0.5f;        // TAU = 2.0
        const float s  = (hv >= 1.0f) ? 1.0f : 0.0f;
        v = hv * (1.0f - s);
        spikes[(size_t)t * BN + idx] = (__bf16)s;
    }
}

// ---------------------------------------------------------------------------
extern "C" void kernel_launch(void* const* d_in, const int* in_sizes, int n_in,
                              void* d_out, int out_size, void* d_ws, size_t ws_size,
                              hipStream_t stream)
{
    const float* x  = (const float*)d_in[0];   // [T, B, N] f32
    const float* w1 = (const float*)d_in[1];   // [N, N] f32
    const float* w2 = (const float*)d_in[2];   // [N, N] f32
    float* out = (float*)d_out;                // [T, B, N] f32

    const int N  = 1024;
    const int K  = N;
    const int T  = 128;
    const int M  = in_sizes[0] / N;            // T*B = 8192
    const int BN = M / T * N;                  // B*N = 65536

    char* ws = (char*)d_ws;
    __bf16* x_bf   = (__bf16*)ws;                       ws += (size_t)M * K * 2;  // 16 MB
    __bf16* w1_bf  = (__bf16*)ws;                       ws += (size_t)N * K * 2;  //  2 MB
    __bf16* w2_bf  = (__bf16*)ws;                       ws += (size_t)N * K * 2;  //  2 MB
    float*  h      = (float*)ws;                        ws += (size_t)M * N * 4;  // 32 MB
    __bf16* spikes = (__bf16*)ws;                                                 // 16 MB

    dim3 block(256);

    // precision prep: everything the matrix units eat is bf16
    cvt_f32_bf16_kernel<<<dim3((M * K / 4 + 255) / 256), block, 0, stream>>>(x,  x_bf,  M * K);
    cvt_f32_bf16_kernel<<<dim3((N * K / 4 + 255) / 256), block, 0, stream>>>(w1, w1_bf, N * K);
    cvt_f32_bf16_kernel<<<dim3((N * K / 4 + 255) / 256), block, 0, stream>>>(w2, w2_bf, N * K);

    dim3 ggrid(N / TILE_N, M / TILE_M);        // (8, 32)

    // fc1: h = x * W1^T
    gemm_nt_bf16_wmma<<<ggrid, block, 0, stream>>>(x_bf, w1_bf, h, M, N, K);
    // LIF temporal scan -> binary spikes (exact in bf16)
    lif_scan_kernel<<<dim3((BN + 255) / 256), block, 0, stream>>>(h, spikes, T, BN);
    // fc2: out = spikes * W2^T
    gemm_nt_bf16_wmma<<<ggrid, block, 0, stream>>>(spikes, w2_bf, out, M, N, K);
}
